// RNN_v2_3599182594141
// MI455X (gfx1250) — compile-verified, matrix operand order checked by reading, not run on hardware
//
#include <hip/hip_runtime.h>
#include <hip/hip_bf16.h>
#include <cstdint>
#include <cstddef>

// Problem dims
#define B_ 16
#define T_ 512
#define V_ 10000
#define E_ 300
#define H_ 256
#define M_ (B_*T_)   // 8192 rows = B*T

typedef __attribute__((ext_vector_type(16))) __bf16        v16bf;
typedef __attribute__((ext_vector_type(8)))  float         v8f;
typedef __attribute__((ext_vector_type(4)))  unsigned int  v4u;
typedef __attribute__((ext_vector_type(8)))  int           v8i;
typedef __attribute__((ext_vector_type(4)))  int           v4i;

union FragBF {
  v4u            q[2];   // two 16-byte chunks
  v16bf          v;      // WMMA operand
  unsigned short s[16];
};
static_assert(sizeof(FragBF) == 32, "frag size");

__device__ __forceinline__ unsigned short f2bf(float f) {
  unsigned u = __float_as_uint(f);
  u += 0x7FFFu + ((u >> 16) & 1u);        // round-to-nearest-even
  return (unsigned short)(u >> 16);
}

// Low 32 bits of a flat pointer into the LDS aperture == LDS byte address
__device__ __forceinline__ unsigned ldsAddr(const void* p) {
  return (unsigned)(uintptr_t)p;
}

// ---- CDNA5 async global->LDS copy (ASYNCcnt tracked), GV addressing mode ----
__device__ __forceinline__ void async_copy_b128(const void* gaddr, unsigned lds_off) {
  asm volatile("global_load_async_to_lds_b128 %0, %1, off"
               :: "v"(lds_off), "v"(gaddr) : "memory");
}
__device__ __forceinline__ void wait_async0() {
  asm volatile("s_wait_asynccnt 0x0" ::: "memory");
}
__device__ __forceinline__ void wait_tensor0() {
#if __has_builtin(__builtin_amdgcn_s_wait_tensorcnt)
  __builtin_amdgcn_s_wait_tensorcnt(0);
#else
  asm volatile("s_wait_tensorcnt 0x0" ::: "memory");
#endif
}

// ---------------------------------------------------------------- converts
__global__ void __launch_bounds__(256)
cvt_f32_to_bf16(const float* __restrict__ src,
                unsigned short* __restrict__ dst, int n) {
  int i = blockIdx.x * blockDim.x + threadIdx.x;
  int stride = gridDim.x * blockDim.x;
  for (; i < n; i += stride) dst[i] = f2bf(src[i]);
}

// ---------------------------------------------------------------- bf16 GEMM
// C[M,N] = A[M,K] * W[N,K]^T   (both K-contiguous row-major)
// 256 threads (8 wave32), tile 128x64, K-step 32, double-buffered LDS with
// async global->LDS staging for bf16 operands (overlapped with WMMA).
#define BM 128
#define BN 64
#define BK 32
#define LDT 40   // padded LDS row stride (bf16) -> conflict-free ds_load_b128

template<bool AF32>
__global__ void __launch_bounds__(256)
gemm_bf16(const void* __restrict__ Av,
          const unsigned short* __restrict__ Bw,
          float* __restrict__ Cf, unsigned short* __restrict__ Cbf,
          int M, int N, int K, int lda, int ldb, int ldc) {
  __shared__ __align__(16) unsigned short Alds[2][8 * 16 * LDT];
  __shared__ __align__(16) unsigned short Blds[2][4 * 16 * LDT];

  const int tid   = threadIdx.x;
  const int lane  = tid & 31;
  const int wid   = tid >> 5;
  const int waveM = wid >> 1;          // 0..3
  const int waveN = wid & 1;           // 0..1
  const int m0    = blockIdx.y * BM;   // M % 128 == 0 for all calls here
  const int n0    = blockIdx.x * BN;

  v8f acc[2][2] = {};

  const float*          Af = (const float*)Av;
  const unsigned short* Ah = (const unsigned short*)Av;

  const int kq  = (tid & 7) * 4;       // 0..28 (scalar staging k quad)
  const int rA  = tid >> 3;            // 0..31 (scalar staging row group)
  const int mlo = lane & 15;
  const int kc  = (lane >> 4) * 8;     // 0 or 8 (frag k-chunk base)
  const bool nfull = (n0 + BN) <= N;

  // Stage one K-slab into buffer `buf`; async where layout allows.
  auto stage = [&](int k0, int buf) {
    const bool kfull = (k0 + BK) <= K;
    // ---- A tile: 128 x 32 ----
    if constexpr (AF32) {
      // fp32 -> bf16 conversion in flight (VALU path; this GEMM is HBM-bound)
      #pragma unroll
      for (int i = 0; i < 4; ++i) {
        int row = rA + 32 * i;
        const float* src = Af + (size_t)(m0 + row) * lda + (k0 + kq);
        union { unsigned short h[4]; uint2 u; } t; t.u = make_uint2(0u, 0u);
        if (kfull) {
          float4 v = *(const float4*)src;
          t.h[0]=f2bf(v.x); t.h[1]=f2bf(v.y); t.h[2]=f2bf(v.z); t.h[3]=f2bf(v.w);
        } else {
          #pragma unroll
          for (int j = 0; j < 4; ++j)
            t.h[j] = (k0 + kq + j < K) ? f2bf(src[j]) : (unsigned short)0;
        }
        int idx = (row >> 4) * (16 * LDT) + (row & 15) * LDT + kq;
        *(uint2*)&Alds[buf][idx] = t.u;
      }
      if (k0 + BK < K)
        __builtin_prefetch(Af + (size_t)(m0 + rA) * lda + (k0 + BK), 0, 1);
    } else {
      if (kfull) {
        // async copy: 512 x 16B chunks, 2 per thread
        #pragma unroll
        for (int i = 0; i < 2; ++i) {
          int p = tid + 256 * i;
          int row = p >> 2, ch = p & 3;                 // 0..127, 0..3
          const unsigned short* g = Ah + (size_t)(m0 + row) * lda + k0 + ch * 8;
          int idx = (row >> 4) * (16 * LDT) + (row & 15) * LDT + ch * 8;
          async_copy_b128(g, ldsAddr(&Alds[buf][idx]));
        }
      } else {
        #pragma unroll
        for (int i = 0; i < 4; ++i) {
          int row = rA + 32 * i;
          const unsigned short* src = Ah + (size_t)(m0 + row) * lda + (k0 + kq);
          union { unsigned short h[4]; uint2 u; } t; t.u = make_uint2(0u, 0u);
          #pragma unroll
          for (int j = 0; j < 4; ++j)
            t.h[j] = (k0 + kq + j < K) ? src[j] : (unsigned short)0;
          int idx = (row >> 4) * (16 * LDT) + (row & 15) * LDT + kq;
          *(uint2*)&Alds[buf][idx] = t.u;
        }
      }
    }
    // ---- B tile: 64 x 32 (always bf16 weights) ----
    if (nfull && kfull) {
      int row = tid >> 2, ch = tid & 3;                 // 0..63, 0..3
      const unsigned short* g = Bw + (size_t)(n0 + row) * ldb + k0 + ch * 8;
      int idx = (row >> 4) * (16 * LDT) + (row & 15) * LDT + ch * 8;
      async_copy_b128(g, ldsAddr(&Blds[buf][idx]));
    } else {
      #pragma unroll
      for (int i = 0; i < 2; ++i) {
        int row = rA + 32 * i;
        int gn  = n0 + row;
        union { unsigned short h[4]; uint2 u; } t; t.u = make_uint2(0u, 0u);
        if (gn < N) {
          const unsigned short* src = Bw + (size_t)gn * ldb + (k0 + kq);
          #pragma unroll
          for (int j = 0; j < 4; ++j)
            t.h[j] = (k0 + kq + j < K) ? src[j] : (unsigned short)0;
        }
        int idx = (row >> 4) * (16 * LDT) + (row & 15) * LDT + kq;
        *(uint2*)&Blds[buf][idx] = t.u;
      }
    }
  };

  int buf = 0;
  stage(0, 0);
  for (int k0 = 0; k0 < K; k0 += BK) {
    wait_async0();            // this wave's async copies into current buf done
    __syncthreads();          // publish all waves' staging (async + ds stores)

    if (k0 + BK < K) stage(k0 + BK, buf ^ 1);   // overlap next slab with WMMA

    FragBF fa[2], fb[2];
    #pragma unroll
    for (int tm = 0; tm < 2; ++tm) {
      int base = (waveM * 2 + tm) * (16 * LDT) + mlo * LDT + kc;
      fa[tm].q[0] = *(const v4u*)&Alds[buf][base];
      fa[tm].q[1] = *(const v4u*)&Alds[buf][base + 16];
    }
    #pragma unroll
    for (int tn = 0; tn < 2; ++tn) {
      int base = (waveN * 2 + tn) * (16 * LDT) + mlo * LDT + kc;
      fb[tn].q[0] = *(const v4u*)&Blds[buf][base];
      fb[tn].q[1] = *(const v4u*)&Blds[buf][base + 16];
    }
    #pragma unroll
    for (int tm = 0; tm < 2; ++tm)
      #pragma unroll
      for (int tn = 0; tn < 2; ++tn)
        acc[tm][tn] = __builtin_amdgcn_wmma_f32_16x16x32_bf16(
            false, fa[tm].v, false, fb[tn].v, (short)0, acc[tm][tn], false, false);
    buf ^= 1;
  }

  // ---- epilogue: C layout VGPR r -> (m=r or r+8 by lane-half, n=lane%16) ----
  const int mrow0 = m0 + waveM * 32;
  const int ncol0 = n0 + waveN * 32;
  #pragma unroll
  for (int tm = 0; tm < 2; ++tm)
    #pragma unroll
    for (int tn = 0; tn < 2; ++tn)
      #pragma unroll
      for (int r = 0; r < 8; ++r) {
        int mm = mrow0 + tm * 16 + r + ((lane >> 4) << 3);
        int nn = ncol0 + tn * 16 + (lane & 15);
        if (nn < N) {
          size_t o = (size_t)mm * ldc + nn;
          if (Cf)  Cf[o]  = acc[tm][tn][r];
          if (Cbf) Cbf[o] = f2bf(acc[tm][tn][r]);
        }
      }
}

// ---------------------------------------------------------------- recurrence
// One persistent workgroup, 16 wave32s. Wave w owns output columns [16w,16w+16).
// W_u B-fragments live in registers for all 512 steps; h lives in LDS as bf16.
// The per-step P tile (16 x 256 fp32, row stride T*H) is prefetched one step
// ahead by the Tensor Data Mover into double-buffered LDS (TENSORcnt tracked).
// __launch_bounds__(512,1): single persistent block -> full register file,
// keeps all 16 fragment sets resident (round-2 build spilled without it).
__global__ void __launch_bounds__(512, 1)
scan_wmma(const float* __restrict__ P, const float* __restrict__ h0,
          const float* __restrict__ Wu, unsigned short* __restrict__ Hbf) {
  __shared__ __align__(16) unsigned short hls[16 * 264];   // padded stride 264
  __shared__ __align__(16) float pls[2][16 * H_];          // TDM double buffer

  const int tid  = threadIdx.x;
  const int lane = tid & 31;
  const int w    = tid >> 5;           // 0..15
  const int mlo  = lane & 15;
  const int kc   = (lane >> 4) * 8;    // 0 or 8
  const int ncol = w * 16 + mlo;       // this lane's output column
  const int mhi  = (lane >> 4) << 3;   // 0 or 8

#if __has_builtin(__builtin_amdgcn_tensor_load_to_lds)
  #define SCAN_TDM 1
#else
  #define SCAN_TDM 0
#endif

#if SCAN_TDM
  // D# group1: data_size=4B(code 2), tensor_dim0=256, tensor_dim1=16,
  // tile_dim0=256, tile_dim1=16, tensor_dim0_stride=T*H=131072 elements.
  const v8i g1 = { 0x00020000, (int)(256u << 16), (int)(16u << 16),
                   (int)(256u << 16), 16, (int)(T_ * H_), 0, 0 };
  const v4i gz = { 0, 0, 0, 0 };
  auto tdm_issue = [&](int t, int buf) {
    unsigned long long ga =
        (unsigned long long)(uintptr_t)(P + (size_t)t * H_);  // tile start
    v4u g0;
    g0[0] = 1u;                                   // count=1, user descriptor
    g0[1] = ldsAddr(&pls[buf][0]);                // LDS byte address
    g0[2] = (unsigned)ga;                         // global_addr[31:0]
    g0[3] = (unsigned)(ga >> 32) | 0x80000000u;   // global_addr[56:32] | type=2
#if defined(__clang_major__) && __clang_major__ >= 23
    const v8i gz8 = { 0, 0, 0, 0, 0, 0, 0, 0 };
    __builtin_amdgcn_tensor_load_to_lds(g0, g1, gz, gz, gz8, 0);
#else
    __builtin_amdgcn_tensor_load_to_lds(g0, g1, gz, gz, 0);
#endif
  };
#endif

  // Preload W_u fragments (fp32 -> bf16 in registers, read once, L2 resident)
  FragBF bfr[8];
  #pragma unroll
  for (int kt = 0; kt < 8; ++kt) {
    const float* src = Wu + (size_t)ncol * H_ + kt * 32 + kc;
    #pragma unroll
    for (int e = 0; e < 8; ++e) bfr[kt].s[e]     = f2bf(src[e]);
    #pragma unroll
    for (int e = 0; e < 8; ++e) bfr[kt].s[8 + e] = f2bf(src[16 + e]);
  }

  // h_0 -> LDS (bf16)
  for (int i = tid; i < 16 * H_; i += 512) {
    int r = i >> 8, c = i & 255;
    hls[r * 264 + c] = f2bf(h0[i]);
  }
#if SCAN_TDM
  if (w == 0) tdm_issue(0, 0);          // prefetch first P tile
#endif
  __syncthreads();

  int buf = 0;
  for (int t = 0; t < T_; ++t) {
#if SCAN_TDM
    if (w == 0) wait_tensor0();         // pls[buf] landed
    __syncthreads();                    // publish TDM tile + prev hls update
#endif
    // A fragments of h from LDS
    FragBF afr[8];
    #pragma unroll
    for (int kt = 0; kt < 8; ++kt) {
      int base = mlo * 264 + kt * 32 + kc;
      afr[kt].q[0] = *(const v4u*)&hls[base];
      afr[kt].q[1] = *(const v4u*)&hls[base + 16];
    }
    // acc starts as p_t tile
    v8f acc = {};
    #pragma unroll
    for (int r = 0; r < 8; ++r) {
      int mm = r + mhi;                                  // batch index b
#if SCAN_TDM
      acc[r] = pls[buf][mm * H_ + ncol];
#else
      acc[r] = P[((size_t)mm * T_ + t) * H_ + ncol];
#endif
    }
#if SCAN_TDM
    if (w == 0 && t + 1 < T_) tdm_issue(t + 1, buf ^ 1);  // overlap next tile
#endif
    #pragma unroll
    for (int kt = 0; kt < 8; ++kt)
      acc = __builtin_amdgcn_wmma_f32_16x16x32_bf16(
          false, afr[kt].v, false, bfr[kt].v, (short)0, acc, false, false);

    // write H_all (bf16) and new h to LDS
    unsigned short hb[8];
    #pragma unroll
    for (int r = 0; r < 8; ++r) {
      int mm = r + mhi;
      hb[r] = f2bf(acc[r]);
      Hbf[((size_t)mm * T_ + t) * H_ + ncol] = hb[r];
    }
    __syncthreads();                                      // all frag reads done
    #pragma unroll
    for (int r = 0; r < 8; ++r) {
      int mm = r + mhi;
      hls[mm * 264 + ncol] = hb[r];
    }
#if !SCAN_TDM
    __syncthreads();
#endif
    buf ^= 1;
  }
}

// ---------------------------------------------------------------- softmax
__device__ __forceinline__ float wave_max32(float v) {
  #pragma unroll
  for (int o = 16; o > 0; o >>= 1) v = fmaxf(v, __shfl_xor(v, o, 32));
  return v;
}
__device__ __forceinline__ float wave_sum32(float v) {
  #pragma unroll
  for (int o = 16; o > 0; o >>= 1) v += __shfl_xor(v, o, 32);
  return v;
}

// One block per row; row (40 KB) staged in LDS -> exactly one global read+write.
__global__ void __launch_bounds__(256)
softmax_rows(float* __restrict__ out) {
  __shared__ float row[V_];
  __shared__ float redm[8];
  __shared__ float reds[8];
  const int tid = threadIdx.x, lane = tid & 31, wid = tid >> 5;
  float* p = out + (size_t)blockIdx.x * V_;

  float mx = -3.402823466e38f;
  for (int i = tid; i < V_; i += 256) { float v = p[i]; row[i] = v; mx = fmaxf(mx, v); }
  mx = wave_max32(mx);
  if (lane == 0) redm[wid] = mx;
  __syncthreads();
  mx = wave_max32((lane < 8) ? redm[lane] : -3.402823466e38f);

  float s = 0.f;
  for (int i = tid; i < V_; i += 256) { float e = __expf(row[i] - mx); row[i] = e; s += e; }
  s = wave_sum32(s);
  if (lane == 0) reds[wid] = s;
  __syncthreads();
  s = wave_sum32((lane < 8) ? reds[lane] : 0.f);

  float inv = 1.0f / s;
  for (int i = tid; i < V_; i += 256) p[i] = row[i] * inv;
}

// ---------------------------------------------------------------- launch
extern "C" void kernel_launch(void* const* d_in, const int* in_sizes, int n_in,
                              void* d_out, int out_size, void* d_ws, size_t ws_size,
                              hipStream_t stream) {
  const float* input  = (const float*)d_in[0];   // [B,T,V]
  const float* hidden = (const float*)d_in[1];   // [B,H]
  const float* W_emb  = (const float*)d_in[2];   // [E,V]
  const float* W_w    = (const float*)d_in[3];   // [H,E]
  const float* W_u    = (const float*)d_in[4];   // [H,H]
  const float* W_v    = (const float*)d_in[5];   // [V,H]
  float* out = (float*)d_out;
  (void)in_sizes; (void)n_in; (void)out_size; (void)ws_size;

  // workspace carve-out (~29 MB total)
  char* base = (char*)d_ws;
  size_t off = 0;
  auto alloc = [&](size_t bytes) -> void* {
    void* p = base + off;
    off = (off + bytes + 255) & ~(size_t)255;
    return p;
  };
  unsigned short* W_emb_bf = (unsigned short*)alloc((size_t)E_ * V_ * 2);
  unsigned short* W_w_bf   = (unsigned short*)alloc((size_t)H_ * E_ * 2);
  unsigned short* W_v_bf   = (unsigned short*)alloc((size_t)V_ * H_ * 2);
  unsigned short* E_bf     = (unsigned short*)alloc((size_t)M_ * E_ * 2);
  float*          Pws      = (float*)alloc((size_t)M_ * H_ * 4);
  unsigned short* H_bf     = (unsigned short*)alloc((size_t)M_ * H_ * 2);

  // weights fp32 -> bf16 (once; they stay L2-resident afterwards)
  cvt_f32_to_bf16<<<512, 256, 0, stream>>>(W_emb, W_emb_bf, E_ * V_);
  cvt_f32_to_bf16<<<64,  256, 0, stream>>>(W_w,   W_w_bf,   H_ * E_);
  cvt_f32_to_bf16<<<512, 256, 0, stream>>>(W_v,   W_v_bf,   V_ * H_);

  // GEMM1: E_all[8192,300] = input @ W_emb^T   (A fp32 converted during staging)
  dim3 g1((E_ + BN - 1) / BN, M_ / BM);
  gemm_bf16<true><<<g1, 256, 0, stream>>>(input, W_emb_bf, nullptr, E_bf,
                                          M_, E_, V_, V_, V_, E_);
  // GEMM2: P_all[8192,256] = E_all @ W_w^T
  dim3 g2((H_ + BN - 1) / BN, M_ / BM);
  gemm_bf16<false><<<g2, 256, 0, stream>>>(E_bf, W_w_bf, Pws, nullptr,
                                           M_, H_, E_, E_, E_, H_);
  // Sequential recurrence (one persistent workgroup, WMMA + TDM per step)
  scan_wmma<<<1, 512, 0, stream>>>(Pws, hidden, W_u, H_bf);

  // GEMM3: logits[8192,10000] = H_all @ W_v^T  -> d_out
  dim3 g3((V_ + BN - 1) / BN, M_ / BM);
  gemm_bf16<false><<<g3, 256, 0, stream>>>(H_bf, W_v_bf, out, nullptr,
                                           M_, V_, H_, H_, H_, V_);
  // Row softmax in place on d_out
  softmax_rows<<<M_, 256, 0, stream>>>(out);
}
